// VectorQuantize_22419729285666
// MI455X (gfx1250) — matrix operand order repaired and checked by reading, not compiled, not run on hardware
//
#include <hip/hip_runtime.h>
#include <hip/hip_bf16.h>

// Problem constants (from reference): x [8,4096,512], embed [8192,512]
#define NN 32768      // rows (8*4096)
#define DD 512        // feature dim
#define KK 8192       // codebook size
#define NTILES (NN / 16)

typedef __bf16 bf16;
typedef __attribute__((ext_vector_type(8)))  __bf16 v8bf;
typedef __attribute__((ext_vector_type(16))) __bf16 v16bf;
typedef __attribute__((ext_vector_type(8)))  float  v8f;
typedef __attribute__((ext_vector_type(8)))  float  f32x8;
typedef __attribute__((ext_vector_type(4)))  float  f32x4;

// LDS B-tile geometry: 16 codes x 512 bf16, padded 8 bf16/row to kill bank conflicts
#define BROW   520                       // elements per LDS row (512 + 8 pad)
#define BROWB  (BROW * 2)                // = 1040 bytes; 260 dwords; 260 % 64 = 4 -> conflict-free
#define BBUFB  (16 * BROWB)              // bytes per buffer = 16640

// ---------------------------------------------------------------------------
// 0) zero a float span in workspace (harness poisons ws with 0xAA)
// ---------------------------------------------------------------------------
__global__ void vq_zero_kernel(float* __restrict__ p, int n) {
    int i = blockIdx.x * blockDim.x + threadIdx.x;
    if (i < n) p[i] = 0.0f;
}

// ---------------------------------------------------------------------------
// 1) prep: embed -> bf16 copy, and embed_sq[k] = sum_d embed[k][d]^2
// ---------------------------------------------------------------------------
__global__ __launch_bounds__(256) void vq_prep_kernel(const float* __restrict__ embed,
                                                      bf16* __restrict__ embB,
                                                      float* __restrict__ embSq) {
    const int k = blockIdx.x;
    const int t = threadIdx.x;
    const float* row = embed + (size_t)k * DD;
    float a = row[t];
    float b = row[t + 256];
    embB[(size_t)k * DD + t]       = (bf16)a;
    embB[(size_t)k * DD + t + 256] = (bf16)b;
    __shared__ float red[256];
    red[t] = a * a + b * b;
    __syncthreads();
    for (int s = 128; s > 0; s >>= 1) {
        if (t < s) red[t] += red[t + s];
        __syncthreads();
    }
    if (t == 0) embSq[k] = red[0];
}

// ---------------------------------------------------------------------------
// 2) fused GEMM + argmax + gather. 8 waves/block, each wave owns a 16-row
//    x-tile; the 16-code B tile is staged once per block into LDS via
//    global_load_async_to_lds_b128 (double-buffered), giving 8x B reuse.
//    dist = 2 * x.Et - ||E||^2 ; argmax over codes, first-max-wins.
// ---------------------------------------------------------------------------
__global__ __launch_bounds__(256) void vq_argmax_kernel(const float* __restrict__ x,
                                                        const bf16*  __restrict__ embB,
                                                        const float* __restrict__ embSq,
                                                        const float* __restrict__ embF,
                                                        float* __restrict__ quantOut,
                                                        float* __restrict__ indOut,
                                                        int*   __restrict__ idxWs) {
    const int wave    = threadIdx.x >> 5;
    const int lane    = threadIdx.x & 31;
    const int tile    = blockIdx.x * 8 + wave;       // grid sized exactly: 0..NTILES-1
    const int rowBase = tile * 16;
    const int half    = lane >> 4;                   // 0: C rows 0-7, 1: C rows 8-15
    const int l16     = lane & 15;

    __shared__ __attribute__((aligned(16))) bf16 ldsB[2][16][BROW];
    const unsigned ldsBase = (unsigned)(size_t)&ldsB[0][0][0];

    // stage code-block cb into LDS buffer buf: 16 rows x 1024B, 16B chunks,
    // 256 threads x 4 chunks, async (ASYNCcnt), no VGPR staging.
    auto stage = [&](int cb, int buf) {
        const char* gbase = (const char*)embB + (size_t)cb * 16 * (DD * 2);
#pragma unroll
        for (int s = 0; s < 4; ++s) {
            const int chunk = threadIdx.x + s * 256;      // 0..1023
            const int row   = chunk >> 6;                 // 64 x 16B chunks per row
            const int col   = chunk & 63;
            const unsigned  ldsOff = ldsBase + (unsigned)(buf * BBUFB + row * BROWB + col * 16);
            const char*     g      = gbase + row * (DD * 2) + col * 16;
            asm volatile("global_load_async_to_lds_b128 %0, %1, off"
                         :: "v"(ldsOff), "v"(g) : "memory");
        }
    };

    // --- load A tile (x rows) as bf16 into registers: 16 chunks of 16x32 ---
    // A k order used here (same in B): lanes 0-15 -> row=lane, K = 0..15
    //                          lanes 16-31 -> row=lane-16, K = 16..31
    v16bf A[16];
    const float* xrow = x + (size_t)(rowBase + l16) * DD;
#pragma unroll
    for (int c = 0; c < 16; ++c) {
        const int dlo = c * 32 + half * 16;
        f32x8 lo = *(const f32x8*)(xrow + dlo);
        f32x8 hi = *(const f32x8*)(xrow + dlo + 8);
        v8bf blo = __builtin_convertvector(lo, v8bf);
        v8bf bhi = __builtin_convertvector(hi, v8bf);
#pragma unroll
        for (int i = 0; i < 8; ++i) { A[c][i] = blo[i]; A[c][i + 8] = bhi[i]; }
    }

    float bestV[8];
    int   bestI[8];
#pragma unroll
    for (int r = 0; r < 8; ++r) { bestV[r] = -3.4e38f; bestI[r] = 0; }

    stage(0, 0);

    // --- stream the codebook: 512 blocks of 16 codes, double-buffered LDS ---
    for (int cb = 0; cb < 512; ++cb) {
        if (cb + 1 < 512) {
            stage(cb + 1, (cb + 1) & 1);
            // 4 new + up to 4 old outstanding; in-order completion =>
            // waiting to <=4 guarantees buffer cb&1 has landed.
            asm volatile("s_wait_asynccnt 0x4" ::: "memory");
        } else {
            asm volatile("s_wait_asynccnt 0x0" ::: "memory");
        }
        __syncthreads();                                 // buffer cb&1 valid for all waves

        const int    cur    = cb & 1;
        const int    myCode = cb * 16 + l16;             // this lane's column
        const bf16*  brow   = &ldsB[cur][l16][0];

        v8f acc = {};
#pragma unroll
        for (int c = 0; c < 16; ++c) {
            // B k order used here (same as A): lane = column,
            // lanes 0-15 hold K=0..15, lanes 16-31 hold K=16..31 (2 per VGPR)
            v16bf B = *(const v16bf*)(brow + c * 32 + half * 16);
            acc = __builtin_amdgcn_wmma_f32_16x16x32_bf16(
                /*neg_a=*/false, A[c], /*neg_b=*/false, B,
                /*c_mod=*/(short)0, acc, /*reuse_a=*/false, /*reuse_b=*/false);
        }
        const float esq = embSq[myCode];
#pragma unroll
        for (int r = 0; r < 8; ++r) {
            float s = 2.0f * acc[r] - esq;               // C/D: half 0 -> M=r, half 1 -> M=r+8
            if (s > bestV[r]) { bestV[r] = s; bestI[r] = myCode; }
        }
        __syncthreads();                                 // done reading before next overwrite
    }

    // --- reduce across the 16 lanes of each half (same rows), smallest idx on ties ---
#pragma unroll
    for (int r = 0; r < 8; ++r) {
#pragma unroll
        for (int off = 8; off >= 1; off >>= 1) {
            float ov = __shfl_xor(bestV[r], off, 16);
            int   oi = __shfl_xor(bestI[r], off, 16);
            if (ov > bestV[r] || (ov == bestV[r] && oi < bestI[r])) {
                bestV[r] = ov; bestI[r] = oi;
            }
        }
    }

    if (l16 == 0) {
#pragma unroll
        for (int r = 0; r < 8; ++r) {
            const int grow = rowBase + half * 8 + r;
            indOut[grow] = (float)bestI[r];
            idxWs[grow]  = bestI[r];
        }
    }

    // --- gather quantize rows: quantize[row] = embed[best_idx[row]] (f32) ---
#pragma unroll
    for (int m = 0; m < 16; ++m) {
        const int src = (m >= 8) ? 16 : 0;
        const int idx = __shfl(bestI[m & 7], src, 32);
        const f32x4* e4 = (const f32x4*)(embF + (size_t)idx * DD);
        f32x4*       q4 = (f32x4*)(quantOut + (size_t)(rowBase + m) * DD);
#pragma unroll
        for (int p = 0; p < 4; ++p) q4[p * 32 + lane] = e4[p * 32 + lane];
    }
}

// ---------------------------------------------------------------------------
// 3) scatter stats: one block per x-row
// ---------------------------------------------------------------------------
__global__ __launch_bounds__(256) void vq_scatter_kernel(const float* __restrict__ x,
                                                         const int* __restrict__ idxWs,
                                                         float* __restrict__ bcs,
                                                         float* __restrict__ esum) {
    const int n = blockIdx.x;
    const int t = threadIdx.x;
    const int idx = idxWs[n];
    if (t == 0) atomicAdd(&bcs[idx], 1.0f);
    const float* xr = x + (size_t)n * DD;
    float* er = esum + (size_t)idx * DD;
    atomicAdd(&er[t],       xr[t]);
    atomicAdd(&er[t + 256], xr[t + 256]);
}

// ---------------------------------------------------------------------------
// 4) EMA blend for embed_avg (elementwise over K*D)
// ---------------------------------------------------------------------------
__global__ __launch_bounds__(256) void vq_ema_kernel(const float* __restrict__ ea,
                                                     const float* __restrict__ esum,
                                                     float* __restrict__ neaOut) {
    const int i = blockIdx.x * 256 + threadIdx.x;
    neaOut[i] = ea[i] * 0.8f + esum[i] * 0.2f;
}

// ---------------------------------------------------------------------------
// 5) per-code stats: new_cluster_size, total, perplexity partial
// ---------------------------------------------------------------------------
__global__ __launch_bounds__(256) void vq_stats_kernel(const float* __restrict__ cs,
                                                       const float* __restrict__ bcs,
                                                       float* __restrict__ ncsOut,
                                                       float* __restrict__ totAcc,
                                                       float* __restrict__ plogAcc) {
    const int k = blockIdx.x * 256 + threadIdx.x;   // K = 8192 -> 32 blocks
    const float b   = bcs[k];
    const float ncs = cs[k] * 0.8f + b * 0.2f;
    ncsOut[k] = ncs;
    atomicAdd(totAcc, ncs);
    const float p = b / (float)NN;
    atomicAdd(plogAcc, p * logf(p + 1e-10f));
}

// ---------------------------------------------------------------------------
// 6) new_embed = new_embed_avg / laplace_smoothed(new_cluster_size) ; perplexity
// ---------------------------------------------------------------------------
__global__ __launch_bounds__(256) void vq_newembed_kernel(const float* __restrict__ neaOut,
                                                          const float* __restrict__ ncsOut,
                                                          const float* __restrict__ totAcc,
                                                          const float* __restrict__ plogAcc,
                                                          float* __restrict__ neOut,
                                                          float* __restrict__ perpOut) {
    const int k = blockIdx.x;
    const int t = threadIdx.x;
    const float total    = *totAcc;
    const float ncs      = ncsOut[k];
    const float smoothed = (ncs + 1e-5f) / (total + (float)KK * 1e-5f) * total;
    const float inv      = 1.0f / smoothed;
    neOut[(size_t)k * DD + t]       = neaOut[(size_t)k * DD + t] * inv;
    neOut[(size_t)k * DD + t + 256] = neaOut[(size_t)k * DD + t + 256] * inv;
    if (k == 0 && t == 0) *perpOut = expf(-(*plogAcc));
}

// ---------------------------------------------------------------------------
extern "C" void kernel_launch(void* const* d_in, const int* in_sizes, int n_in,
                              void* d_out, int out_size, void* d_ws, size_t ws_size,
                              hipStream_t stream) {
    const float* x     = (const float*)d_in[0];   // [N, D]
    const float* embed = (const float*)d_in[1];   // [K, D]
    const float* cs    = (const float*)d_in[2];   // [K]
    const float* ea    = (const float*)d_in[3];   // [K, D]

    // d_out layout: quantize | embed_ind | new_cluster_size | new_embed_avg | new_embed | perplexity
    float* out     = (float*)d_out;
    float* quant   = out;                               // N*D   = 16777216
    float* indOut  = quant  + (size_t)NN * DD;          // N     =    32768
    float* ncsOut  = indOut + NN;                       // K     =     8192
    float* neaOut  = ncsOut + KK;                       // K*D   =  4194304
    float* neOut   = neaOut + (size_t)KK * DD;          // K*D   =  4194304
    float* perpOut = neOut  + (size_t)KK * DD;          // 1

    // workspace layout
    char*  ws     = (char*)d_ws;
    bf16*  embB   = (bf16*)ws;                                      // K*D*2 = 8 MB
    size_t off    = (size_t)KK * DD * sizeof(bf16);
    float* embSq  = (float*)(ws + off);  off += (size_t)KK * sizeof(float);
    float* bcs    = (float*)(ws + off);  off += (size_t)KK * sizeof(float);
    float* esum   = (float*)(ws + off);  off += (size_t)KK * DD * sizeof(float);
    int*   idxWs  = (int*)  (ws + off);  off += (size_t)NN * sizeof(int);
    float* accs   = (float*)(ws + off);  // [0]=total, [1]=sum p*log(p)

    // zero: bcs .. esum .. idxWs .. accs (contiguous span)
    const int nz = KK + KK * DD + NN + 2;
    vq_zero_kernel<<<(nz + 255) / 256, 256, 0, stream>>>(bcs, nz);

    vq_prep_kernel<<<KK, 256, 0, stream>>>(embed, embB, embSq);

    vq_argmax_kernel<<<NTILES / 8, 256, 0, stream>>>(x, embB, embSq, embed,
                                                     quant, indOut, idxWs);

    vq_scatter_kernel<<<NN, 256, 0, stream>>>(x, idxWs, bcs, esum);

    vq_ema_kernel<<<(KK * DD) / 256, 256, 0, stream>>>(ea, esum, neaOut);

    vq_stats_kernel<<<KK / 256, 256, 0, stream>>>(cs, bcs, ncsOut, accs, accs + 1);

    vq_newembed_kernel<<<KK, 256, 0, stream>>>(neaOut, ncsOut, accs, accs + 1,
                                               neOut, perpOut);
}
